// GCN_Top_63290638074050
// MI455X (gfx1250) — compile-verified, matrix operand order checked by reading, not run on hardware
//
#include <hip/hip_runtime.h>
#include <hip/hip_bf16.h>

#define HID 512
#define NTILES 32          // HID / 16

typedef __attribute__((ext_vector_type(8)))  __bf16 v8bf;
typedef __attribute__((ext_vector_type(16))) __bf16 v16bf;
typedef __attribute__((ext_vector_type(8)))  float  v8f;

__device__ __forceinline__ unsigned short f2bf(float f) {
    unsigned int u = __float_as_uint(f);
    unsigned int r = u + 0x7FFFu + ((u >> 16) & 1u);   // round-to-nearest-even
    return (unsigned short)(r >> 16);
}

// ---------------------------------------------------------------- elementwise
__global__ void k_cvt_bf16(const float* __restrict__ x, unsigned short* __restrict__ xb,
                           long long total) {
    long long i = (long long)blockIdx.x * blockDim.x + threadIdx.x;
    if (i < total) xb[i] = f2bf(x[i]);
}

__global__ void k_deg_init(int* deg, int n) {
    int i = blockIdx.x * blockDim.x + threadIdx.x;
    if (i < n) deg[i] = 1;                       // self-loop
}

__global__ void k_deg_edges(int* deg, const int* __restrict__ dst, int ne) {
    int e = blockIdx.x * blockDim.x + threadIdx.x;
    if (e < ne) atomicAdd(&deg[dst[e]], 1);
}

__global__ void k_dis(float* __restrict__ dis, const int* __restrict__ deg, int n) {
    int i = blockIdx.x * blockDim.x + threadIdx.x;
    if (i < n) dis[i] = rsqrtf((float)deg[i]);
}

__global__ void k_norm(float* __restrict__ nrm, const float* __restrict__ dis,
                       const int* __restrict__ src, const int* __restrict__ dst, int ne) {
    int e = blockIdx.x * blockDim.x + threadIdx.x;
    if (e < ne) nrm[e] = dis[src[e]] * dis[dst[e]];
}

// ---------------------------------------------------------------- W packing
// Pack W[K][HID] (f32 row-major) into per-wave WMMA B fragments:
// Wp[((kt*NTILES + nt)*32 + lane)*16 + e], lane holds col nt*16+(lane&15),
// K index = kt*32 + (lane>>4)*8 + (e<8 ? e : e+8)   (16-bit B layout, wave32)
__global__ void k_pack_w(const float* __restrict__ W, unsigned short* __restrict__ Wp, int K) {
    int idx = blockIdx.x * blockDim.x + threadIdx.x;     // = tile*32 + lane
    int total = (K >> 5) * NTILES * 32;
    if (idx >= total) return;
    int lane = idx & 31;
    int tile = idx >> 5;
    int nt = tile % NTILES;
    int kt = tile / NTILES;
    int col = nt * 16 + (lane & 15);
    int kb  = kt * 32 + ((lane >> 4) << 3);
#pragma unroll
    for (int e = 0; e < 16; ++e) {
        int kk = kb + (e < 8 ? e : e + 8);
        Wp[(size_t)idx * 16 + e] = f2bf(W[(size_t)kk * HID + col]);
    }
}

// ---------------------------------------------------------------- WMMA GEMM (+ fused self-loop init)
// C[M][HID] = A[M][K] (bf16) @ Bp (packed bf16), f32 accum.
// Epilogue also writes acc = dis[row]^2 * C  (self-loop term of the GCN aggregation).
// Block = 8 waves; wave w computes rows [blockIdx.y*128 + w*16, +16), cols [blockIdx.x*64, +64).
__global__ void __launch_bounds__(256)
k_gemm(const unsigned short* __restrict__ A, const unsigned short* __restrict__ Bp,
       float* __restrict__ C, float* __restrict__ acc, const float* __restrict__ dis,
       int M, int K) {
    int wave = threadIdx.x >> 5;
    int lane = threadIdx.x & 31;
    int m0 = (blockIdx.y * 8 + wave) * 16;
    if (m0 >= M) return;                          // wave-uniform: EXEC stays all-ones
    int n0 = blockIdx.x * 64;
    int row  = m0 + (lane & 15);
    int koff = (lane >> 4) << 3;                  // 0 or 8

    v8f acc0 = {}, acc1 = {}, acc2 = {}, acc3 = {};
    int kTiles = K >> 5;
    const int nt0 = n0 >> 4;

    for (int kt = 0; kt < kTiles; ++kt) {
        // A fragment: two contiguous 16B loads per lane
        const unsigned short* ap = A + (size_t)row * K + kt * 32 + koff;
        union { v16bf v; v8bf h[2]; } af;
        af.h[0] = *(const v8bf*)(ap);
        af.h[1] = *(const v8bf*)(ap + 16);

        // B fragments: contiguous 32B per lane, 512-element stride between n-tiles
        const unsigned short* bp = Bp + (((size_t)kt * NTILES + nt0) * 32 + lane) * 16;
        v16bf b0 = *(const v16bf*)(bp);
        v16bf b1 = *(const v16bf*)(bp + 512);
        v16bf b2 = *(const v16bf*)(bp + 1024);
        v16bf b3 = *(const v16bf*)(bp + 1536);

        acc0 = __builtin_amdgcn_wmma_f32_16x16x32_bf16(false, af.v, false, b0, (short)0, acc0, false, false);
        acc1 = __builtin_amdgcn_wmma_f32_16x16x32_bf16(false, af.v, false, b1, (short)0, acc1, false, false);
        acc2 = __builtin_amdgcn_wmma_f32_16x16x32_bf16(false, af.v, false, b2, (short)0, acc2, false, false);
        acc3 = __builtin_amdgcn_wmma_f32_16x16x32_bf16(false, af.v, false, b3, (short)0, acc3, false, false);
    }

    // D layout: VGPR v, lanes 0-15 -> row m0+v, lanes 16-31 -> row m0+v+8; col = n0 + (lane&15)
    int rbase = m0 + ((lane >> 4) << 3);
    int cbase = n0 + (lane & 15);
#pragma unroll
    for (int v = 0; v < 8; ++v) {
        int rr = rbase + v;
        float d = dis[rr];                         // uniform per half-wave -> broadcast
        float s = d * d;
        size_t r = (size_t)rr * HID;
        C[r + cbase]        = acc0[v];
        C[r + cbase + 16]   = acc1[v];
        C[r + cbase + 32]   = acc2[v];
        C[r + cbase + 48]   = acc3[v];
        acc[r + cbase]      = acc0[v] * s;
        acc[r + cbase + 16] = acc1[v] * s;
        acc[r + cbase + 32] = acc2[v] * s;
        acc[r + cbase + 48] = acc3[v] * s;
    }
}

// ---------------------------------------------------------------- aggregation
__global__ void k_scatter(float* __restrict__ acc, const float* __restrict__ t,
                          const float* __restrict__ nrm,
                          const int* __restrict__ src, const int* __restrict__ dst, int ne) {
    long long idx = (long long)blockIdx.x * blockDim.x + threadIdx.x;
    int e = (int)(idx >> 7);                      // 128 threads per edge
    if (e >= ne) return;
    int c = (int)(idx & 127);
    int s = src[e], d = dst[e];
    float w = nrm[e];
    float4 v = *(const float4*)(t + (size_t)s * HID + c * 4);
    float* p = acc + (size_t)d * HID + c * 4;
    unsafeAtomicAdd(p + 0, v.x * w);
    unsafeAtomicAdd(p + 1, v.y * w);
    unsafeAtomicAdd(p + 2, v.z * w);
    unsafeAtomicAdd(p + 3, v.w * w);
}

__global__ void k_bias_tanh(float* __restrict__ acc, unsigned short* __restrict__ hbf,
                            const float* __restrict__ bias, long long total) {
    long long i = (long long)blockIdx.x * blockDim.x + threadIdx.x;
    if (i >= total) return;
    int f = (int)(i & (HID - 1));
    float v = tanhf(acc[i] + bias[f]);
    acc[i] = v;
    hbf[i] = f2bf(v);
}

// ---------------------------------------------------------------- final edge MLP
__global__ void k_final(const float* __restrict__ h3, const int* __restrict__ ei,
                        const int* __restrict__ trainId, const float* __restrict__ fcW,
                        const float* __restrict__ fcb, float* __restrict__ out,
                        int nTrain, int nE) {
    int t = blockIdx.x;
    if (t >= nTrain) return;
    int e = trainId[t];
    int a = ei[e];
    int b = ei[nE + e];
    float s = 0.f;
    for (int f = threadIdx.x; f < HID; f += blockDim.x)
        s += h3[(size_t)a * HID + f] * h3[(size_t)b * HID + f] * fcW[f];
    __shared__ float red[4];
#pragma unroll
    for (int o = 16; o > 0; o >>= 1) s += __shfl_down(s, o, 32);
    if ((threadIdx.x & 31) == 0) red[threadIdx.x >> 5] = s;
    __syncthreads();
    if (threadIdx.x == 0)
        out[t] = red[0] + red[1] + red[2] + red[3] + fcb[0];
}

// ---------------------------------------------------------------- launch
static inline size_t alignup(size_t x) { return (x + 255) & ~(size_t)255; }
static inline int nblk(long long total, int bs) { return (int)((total + bs - 1) / bs); }

extern "C" void kernel_launch(void* const* d_in, const int* in_sizes, int n_in,
                              void* d_out, int out_size, void* d_ws, size_t ws_size,
                              hipStream_t stream) {
    const float* x   = (const float*)d_in[0];
    const int*   ei  = (const int*)  d_in[1];
    const int*   tri = (const int*)  d_in[2];
    const float* W1  = (const float*)d_in[3];
    const float* b1  = (const float*)d_in[4];
    const float* W2  = (const float*)d_in[5];
    const float* b2  = (const float*)d_in[6];
    const float* W3  = (const float*)d_in[7];
    const float* b3  = (const float*)d_in[8];
    const float* fcW = (const float*)d_in[9];
    const float* fcb = (const float*)d_in[10];
    float* out = (float*)d_out;

    const int inDim = in_sizes[3] / HID;          // 128
    const int N  = in_sizes[0] / inDim;           // 50000
    const int nE = in_sizes[1] / 2;               // 400000
    const int nT = in_sizes[2];                   // 50000
    const int* srcp = ei;
    const int* dstp = ei + nE;

    char* ws = (char*)d_ws;
    size_t off = 0;
    auto take = [&](size_t bytes) { void* p = ws + off; off = alignup(off + bytes); return p; };
    float*          t    = (float*)          take((size_t)N * HID * 4);
    float*          acc  = (float*)          take((size_t)N * HID * 4);
    unsigned short* inbf = (unsigned short*) take((size_t)N * HID * 2);
    unsigned short* w1p  = (unsigned short*) take((size_t)inDim * HID * 2);
    unsigned short* w2p  = (unsigned short*) take((size_t)HID * HID * 2);
    unsigned short* w3p  = (unsigned short*) take((size_t)HID * HID * 2);
    int*            deg  = (int*)            take((size_t)N * 4);
    float*          dis  = (float*)          take((size_t)N * 4);
    float*          nrm  = (float*)          take((size_t)nE * 4);

    // --- graph preprocessing
    k_deg_init<<<nblk(N, 256), 256, 0, stream>>>(deg, N);
    k_deg_edges<<<nblk(nE, 256), 256, 0, stream>>>(deg, dstp, nE);
    k_dis<<<nblk(N, 256), 256, 0, stream>>>(dis, deg, N);
    k_norm<<<nblk(nE, 256), 256, 0, stream>>>(nrm, dis, srcp, dstp, nE);

    // --- weight packing + input conversion
    k_pack_w<<<nblk((inDim >> 5) * NTILES * 32, 256), 256, 0, stream>>>(W1, w1p, inDim);
    k_pack_w<<<nblk((HID   >> 5) * NTILES * 32, 256), 256, 0, stream>>>(W2, w2p, HID);
    k_pack_w<<<nblk((HID   >> 5) * NTILES * 32, 256), 256, 0, stream>>>(W3, w3p, HID);
    k_cvt_bf16<<<nblk((long long)N * inDim, 256), 256, 0, stream>>>(x, inbf, (long long)N * inDim);

    const long long totElem = (long long)N * HID;
    dim3 gemmGrid(HID / 64, (N + 127) / 128);

    // --- layer 1
    k_gemm<<<gemmGrid, 256, 0, stream>>>(inbf, w1p, t, acc, dis, N, inDim);
    k_scatter<<<nblk((long long)nE * 128, 256), 256, 0, stream>>>(acc, t, nrm, srcp, dstp, nE);
    k_bias_tanh<<<nblk(totElem, 256), 256, 0, stream>>>(acc, inbf, b1, totElem);

    // --- layer 2
    k_gemm<<<gemmGrid, 256, 0, stream>>>(inbf, w2p, t, acc, dis, N, HID);
    k_scatter<<<nblk((long long)nE * 128, 256), 256, 0, stream>>>(acc, t, nrm, srcp, dstp, nE);
    k_bias_tanh<<<nblk(totElem, 256), 256, 0, stream>>>(acc, inbf, b2, totElem);

    // --- layer 3
    k_gemm<<<gemmGrid, 256, 0, stream>>>(inbf, w3p, t, acc, dis, N, HID);
    k_scatter<<<nblk((long long)nE * 128, 256), 256, 0, stream>>>(acc, t, nrm, srcp, dstp, nE);
    k_bias_tanh<<<nblk(totElem, 256), 256, 0, stream>>>(acc, inbf, b3, totElem);

    // --- final edge scoring (acc now holds h3 in f32)
    k_final<<<nT, 128, 0, stream>>>(acc, ei, tri, fcW, fcb, out, nT, nE);
}
